// PhysicsGuidedSparseAttention_81277961109854
// MI455X (gfx1250) — matrix operand order repaired and checked by reading, not compiled, not run on hardware
//
#include <hip/hip_runtime.h>
#include <stdint.h>

// ---------------- CDNA5 WMMA types ----------------
typedef __attribute__((ext_vector_type(16))) __bf16 v16bf;
typedef __attribute__((ext_vector_type(8)))  float  v8f;

#define DIM    256
#define NHEAD  8
#define HDIM   32
#define NTOK   4096
#define QKVN   768
#define ATT_SCALE 0.17677669529663687f   // 32^-0.5

union frag16 { v16bf bf; unsigned u[8]; };

__device__ __forceinline__ unsigned short f2bf(float f) {
  union { float f; unsigned u; } cv; cv.f = f;
  unsigned u = cv.u;
  u += 0x7FFFu + ((u >> 16) & 1u);   // round-to-nearest-even
  return (unsigned short)(u >> 16);
}

// ---------------- keep mask: 3x3 dilation of (dbz >= 15) ----------------
__global__ void keep_kernel(const float* __restrict__ dbz, float* __restrict__ keep) {
  int t = blockIdx.x * blockDim.x + threadIdx.x;
  if (t >= NTOK) return;
  int frame = t >> 10;          // 32x32 frames
  int pos   = t & 1023;
  int i = pos >> 5, j = pos & 31;
  const float* base = dbz + frame * 1024;
  float mx = 0.0f;
  #pragma unroll
  for (int di = -1; di <= 1; ++di) {
    int ii = i + di; if (ii < 0 || ii > 31) continue;
    #pragma unroll
    for (int dj = -1; dj <= 1; ++dj) {
      int jj = j + dj; if (jj < 0 || jj > 31) continue;
      if (base[ii * 32 + jj] >= 15.0f) mx = 1.0f;
    }
  }
  keep[t] = mx;
}

// ---------------- f32 -> bf16 convert ----------------
__global__ void cvt_kernel(const float* __restrict__ src, unsigned short* __restrict__ dst, int n) {
  int t = blockIdx.x * blockDim.x + threadIdx.x;
  if (t < n) dst[t] = f2bf(src[t]);
}

// ---------------- QKV GEMM: X(4096x256) @ Wqkv^T -> Q,K (h,n,d) and V^T (h,d,n) ----------------
__global__ __launch_bounds__(256) void qkv_gemm(const unsigned short* __restrict__ X,
                                                const unsigned short* __restrict__ W,
                                                unsigned short* __restrict__ Q,
                                                unsigned short* __restrict__ K,
                                                unsigned short* __restrict__ Vt) {
  int wave = (int)((blockIdx.x * blockDim.x + threadIdx.x) >> 5);
  int lane = threadIdx.x & 31;
  int mtile = wave / 48;            // 256 M-tiles
  int ntile = wave % 48;            // 48 N-tiles (768 cols)
  int g  = lane >> 4;               // half-wave group
  int ln = lane & 15;

  const unsigned* arow = (const unsigned*)(X + (size_t)(mtile * 16 + ln) * DIM);
  const unsigned* brow = (const unsigned*)(W + (size_t)(ntile * 16 + ln) * DIM);

  v8f acc = {};
  #pragma unroll
  for (int kt = 0; kt < 8; ++kt) {
    int kb0 = kt * 32;
    frag16 a, b;
    #pragma unroll
    for (int v = 0; v < 8; ++v) {
      int ka = kb0 + (g << 4) + (v << 1);                           // A layout
      int kb = kb0 + (g << 4) + (v << 1);                           // B layout
      a.u[v] = arow[ka >> 1];
      b.u[v] = brow[kb >> 1];
    }
    acc = __builtin_amdgcn_wmma_f32_16x16x32_bf16(false, a.bf, false, b.bf,
                                                  (short)0, acc, false, false);
  }

  int j = ntile * 16 + ln;          // output column 0..767
  int s = j >> 8;                   // 0=Q 1=K 2=V
  int h = (j & 255) >> 5;
  int d = j & 31;
  #pragma unroll
  for (int r = 0; r < 8; ++r) {
    int tok = mtile * 16 + r + (g << 3);
    unsigned short val = f2bf(acc[r]);
    if (s == 0)      Q [((size_t)h * NTOK + tok) * HDIM + d] = val;
    else if (s == 1) K [((size_t)h * NTOK + tok) * HDIM + d] = val;
    else             Vt[((size_t)h * HDIM + d) * NTOK + tok] = val;
  }
}

// ---------------- Flash attention: one wave per (head, 16-query tile) ----------------
__global__ __launch_bounds__(256) void attn_kernel(const unsigned short* __restrict__ Q,
                                                   const unsigned short* __restrict__ K,
                                                   const unsigned short* __restrict__ Vt,
                                                   const float* __restrict__ keep,
                                                   unsigned short* __restrict__ O) {
  __shared__ unsigned short pbuf[8][16 * 32];   // per-wave P bounce buffer (bf16)
  int wslot = threadIdx.x >> 5;
  int lane  = threadIdx.x & 31;
  int wave  = blockIdx.x * 8 + wslot;
  int h  = wave >> 8;                 // 8 heads
  int qt = wave & 255;                // 256 query tiles
  int g  = lane >> 4;
  int ln = lane & 15;
  unsigned short* pb = &pbuf[wslot][0];

  // Q fragment (A layout), loaded once
  frag16 qf;
  {
    const unsigned* qrow = (const unsigned*)(Q + ((size_t)h * NTOK + qt * 16 + ln) * HDIM);
    #pragma unroll
    for (int v = 0; v < 8; ++v) {
      int ka = (g << 4) + (v << 1);
      qf.u[v] = qrow[ka >> 1];
    }
  }

  v8f o0 = {}, o1 = {};
  float mrun[8], lrun[8];
  #pragma unroll
  for (int r = 0; r < 8; ++r) { mrun[r] = -3.0e38f; lrun[r] = 0.0f; }

  for (int c = 0; c < 128; ++c) {     // 128 chunks of 32 keys
    int k0 = c * 32;
    // --- S = Q K^T for two 16-key subtiles ---
    frag16 kf0, kf1;
    const unsigned* kr0 = (const unsigned*)(K + ((size_t)h * NTOK + k0 + ln) * HDIM);
    const unsigned* kr1 = (const unsigned*)(K + ((size_t)h * NTOK + k0 + 16 + ln) * HDIM);
    #pragma unroll
    for (int v = 0; v < 8; ++v) {
      int kb = (g << 4) + (v << 1);
      kf0.u[v] = kr0[kb >> 1];
      kf1.u[v] = kr1[kb >> 1];
    }
    v8f z = {};
    v8f s0 = __builtin_amdgcn_wmma_f32_16x16x32_bf16(false, qf.bf, false, kf0.bf, (short)0, z, false, false);
    v8f s1 = __builtin_amdgcn_wmma_f32_16x16x32_bf16(false, qf.bf, false, kf1.bf, (short)0, z, false, false);

    bool kp0 = keep[k0 + ln]      > 0.5f;
    bool kp1 = keep[k0 + 16 + ln] > 0.5f;

    // --- online softmax (row stats across 16-lane halves) ---
    float a0[8], a1[8];
    #pragma unroll
    for (int r = 0; r < 8; ++r) {
      a0[r] = kp0 ? s0[r] * ATT_SCALE : -3.0e38f;
      a1[r] = kp1 ? s1[r] * ATT_SCALE : -3.0e38f;
      float mx = fmaxf(a0[r], a1[r]);
      mx = fmaxf(mx, __shfl_xor(mx, 1));
      mx = fmaxf(mx, __shfl_xor(mx, 2));
      mx = fmaxf(mx, __shfl_xor(mx, 4));
      mx = fmaxf(mx, __shfl_xor(mx, 8));
      float mnew  = fmaxf(mrun[r], mx);
      float alpha = __expf(mrun[r] - mnew);
      float e0 = kp0 ? __expf(a0[r] - mnew) : 0.0f;
      float e1 = kp1 ? __expf(a1[r] - mnew) : 0.0f;
      float sum = e0 + e1;
      sum += __shfl_xor(sum, 1);
      sum += __shfl_xor(sum, 2);
      sum += __shfl_xor(sum, 4);
      sum += __shfl_xor(sum, 8);
      lrun[r] = lrun[r] * alpha + sum;
      mrun[r] = mnew;
      o0[r] = o0[r] * alpha;
      o1[r] = o1[r] * alpha;
      int m = r + (g << 3);
      pb[m * 32 + ln]      = f2bf(e0);    // C-layout -> LDS, row-major 16x32
      pb[m * 32 + 16 + ln] = f2bf(e1);
    }

    // make the whole P tile visible, then regather in A layout
    __builtin_amdgcn_wave_barrier();
    asm volatile("s_wait_dscnt 0x0" ::: "memory");
    frag16 pf;
    const unsigned* prow = (const unsigned*)(pb + ln * 32);
    #pragma unroll
    for (int v = 0; v < 8; ++v) {
      int ka = (g << 4) + (v << 1);
      pf.u[v] = prow[ka >> 1];
    }
    __builtin_amdgcn_wave_barrier();
    asm volatile("" ::: "memory");      // keep next-iter LDS stores below the loads

    // --- O += P @ V (two d-halves) ---
    frag16 vf0, vf1;
    const unsigned* vr0 = (const unsigned*)(Vt + ((size_t)h * HDIM + ln) * NTOK + k0);
    const unsigned* vr1 = (const unsigned*)(Vt + ((size_t)h * HDIM + 16 + ln) * NTOK + k0);
    #pragma unroll
    for (int v = 0; v < 8; ++v) {
      int kb = (g << 4) + (v << 1);
      vf0.u[v] = vr0[kb >> 1];
      vf1.u[v] = vr1[kb >> 1];
    }
    o0 = __builtin_amdgcn_wmma_f32_16x16x32_bf16(false, pf.bf, false, vf0.bf, (short)0, o0, false, false);
    o1 = __builtin_amdgcn_wmma_f32_16x16x32_bf16(false, pf.bf, false, vf1.bf, (short)0, o1, false, false);
  }

  // normalize + store (n, h*32+d) bf16
  #pragma unroll
  for (int r = 0; r < 8; ++r) {
    float inv = (lrun[r] > 0.0f) ? (1.0f / lrun[r]) : 0.0f;
    int tok = qt * 16 + r + (g << 3);
    O[(size_t)tok * DIM + h * HDIM + ln]      = f2bf(o0[r] * inv);
    O[(size_t)tok * DIM + h * HDIM + 16 + ln] = f2bf(o1[r] * inv);
  }
}

// ---------------- proj GEMM + bias + query gating + residual ----------------
__global__ __launch_bounds__(256) void proj_kernel(const unsigned short* __restrict__ O,
                                                   const unsigned short* __restrict__ Wp,
                                                   const float* __restrict__ bias,
                                                   const float* __restrict__ x,
                                                   const float* __restrict__ keep,
                                                   float* __restrict__ out) {
  int wave = (int)((blockIdx.x * blockDim.x + threadIdx.x) >> 5);
  int lane = threadIdx.x & 31;
  int mtile = wave >> 4;            // 256
  int ntile = wave & 15;            // 16
  int g  = lane >> 4;
  int ln = lane & 15;

  const unsigned* arow = (const unsigned*)(O  + (size_t)(mtile * 16 + ln) * DIM);
  const unsigned* brow = (const unsigned*)(Wp + (size_t)(ntile * 16 + ln) * DIM);

  v8f acc = {};
  #pragma unroll
  for (int kt = 0; kt < 8; ++kt) {
    int kb0 = kt * 32;
    frag16 a, b;
    #pragma unroll
    for (int v = 0; v < 8; ++v) {
      int ka = kb0 + (g << 4) + (v << 1);
      int kb = kb0 + (g << 4) + (v << 1);
      a.u[v] = arow[ka >> 1];
      b.u[v] = brow[kb >> 1];
    }
    acc = __builtin_amdgcn_wmma_f32_16x16x32_bf16(false, a.bf, false, b.bf,
                                                  (short)0, acc, false, false);
  }

  int j = ntile * 16 + ln;
  float bj = bias[j];
  #pragma unroll
  for (int r = 0; r < 8; ++r) {
    int tok = mtile * 16 + r + (g << 3);
    size_t idx = (size_t)tok * DIM + j;
    float y = (keep[tok] > 0.5f) ? (acc[r] + bj) : 0.0f;
    out[idx] = x[idx] + y;
  }
}

// ---------------- host ----------------
extern "C" void kernel_launch(void* const* d_in, const int* in_sizes, int n_in,
                              void* d_out, int out_size, void* d_ws, size_t ws_size,
                              hipStream_t stream) {
  const float* x      = (const float*)d_in[0];   // 4096*256
  const float* dbz    = (const float*)d_in[1];   // 4096
  const float* qkv_w  = (const float*)d_in[2];   // 768*256
  const float* proj_w = (const float*)d_in[3];   // 256*256
  const float* proj_b = (const float*)d_in[4];   // 256
  float* out = (float*)d_out;

  char* ws = (char*)d_ws;
  size_t off = 0;
  auto carve = [&](size_t bytes) -> void* {
    void* p = ws + off;
    off += (bytes + 255) & ~(size_t)255;
    return p;
  };
  unsigned short* X16  = (unsigned short*)carve((size_t)NTOK * DIM * 2);
  unsigned short* Wq16 = (unsigned short*)carve((size_t)QKVN * DIM * 2);
  unsigned short* Wp16 = (unsigned short*)carve((size_t)DIM * DIM * 2);
  float*          keep = (float*)carve((size_t)NTOK * 4);
  unsigned short* Qb   = (unsigned short*)carve((size_t)NHEAD * NTOK * HDIM * 2);
  unsigned short* Kb   = (unsigned short*)carve((size_t)NHEAD * NTOK * HDIM * 2);
  unsigned short* Vt   = (unsigned short*)carve((size_t)NHEAD * NTOK * HDIM * 2);
  unsigned short* O16  = (unsigned short*)carve((size_t)NTOK * DIM * 2);

  cvt_kernel<<<(NTOK * DIM + 255) / 256, 256, 0, stream>>>(x, X16, NTOK * DIM);
  cvt_kernel<<<(QKVN * DIM + 255) / 256, 256, 0, stream>>>(qkv_w, Wq16, QKVN * DIM);
  cvt_kernel<<<(DIM * DIM + 255) / 256, 256, 0, stream>>>(proj_w, Wp16, DIM * DIM);
  keep_kernel<<<NTOK / 256, 256, 0, stream>>>(dbz, keep);

  qkv_gemm<<<(256 * 48) / 8, 256, 0, stream>>>(X16, Wq16, Qb, Kb, Vt);   // 1536 blocks
  attn_kernel<<<(NHEAD * 256) / 8, 256, 0, stream>>>(Qb, Kb, Vt, keep, O16); // 256 blocks
  proj_kernel<<<(256 * 16) / 8, 256, 0, stream>>>(O16, Wp16, proj_b, x, keep, out); // 512 blocks
}
